// WorldModel_17875653886399
// MI455X (gfx1250) — matrix operand order repaired and checked by reading, not compiled, not run on hardware
//
#include <hip/hip_runtime.h>
#include <math.h>

// ---------------------------------------------------------------------------
// World-model rollout: 127 sequential steps of (MLP 16->512->512->512->6,
// leaky ReLU) + rigid-body RK4.  GEMMs via V_WMMA_F32_16X16X32_F16.
// Persistent kernel: 64 blocks x 32 batch rows, whole scan inside one launch.
// ---------------------------------------------------------------------------

constexpr int   kB      = 2048;
constexpr int   kState  = 12;
constexpr int   kActD   = 4;
constexpr int   kT      = 128;
constexpr int   kTF     = kT - 1;     // 127 force steps
constexpr int   kH      = 512;
constexpr int   kM      = 32;         // batch rows per block
constexpr int   kWaves  = 8;          // 256 threads
constexpr float kDT     = 0.01f;
constexpr float kInvMass = 1.0f / 1.5f;
constexpr float kEps    = 1e-6f;

// f16 workspace offsets (elements)
constexpr size_t kW0T = 0;                    // [512][32]  (K padded 16->32)
constexpr size_t kW1T = kW0T + 512 * 32;      // [512][512]
constexpr size_t kW2T = kW1T + 512 * 512;     // [512][512]
constexpr size_t kW3T = kW2T + 512 * 512;     // [16][512]  (N padded 6->16)

// LDS layout
constexpr int kInBufH = kM * 32;              // f16, layer-0 input (K padded)
constexpr int kBufH   = kM * kH;              // f16, activation ping/pong
constexpr int kFnF    = kM * 16;              // f32, MLP head output
constexpr int kXsF    = kM * kState;          // f32, persistent state
constexpr int kSmemBytes = (kInBufH + 2 * kBufH) * 2 + (kFnF + kXsF) * 4;

typedef __attribute__((ext_vector_type(16))) _Float16 v16h;
typedef __attribute__((ext_vector_type(8)))  _Float16 v8h;
typedef __attribute__((ext_vector_type(8)))  float    v8f;

union H16 { v16h v; v8h h[2]; };

// A operand 16x32 f16 (row-major source, stride in f16 elements).
// lane<16: M=lane, K={0..7}u{16..23}; lane>=16: M=lane-16, K={8..15}u{24..31}.
__device__ __forceinline__ v16h load_a_tile(const _Float16* __restrict__ base,
                                            int stride, int lane) {
  const int r    = lane & 15;
  const int half = lane >> 4;
  const _Float16* p = base + r * stride + half * 8;
  H16 u;
  u.h[0] = *(const v8h*)(p);
  u.h[1] = *(const v8h*)(p + 16);
  return u.v;
}

// B operand 32x16 f16 from N-major (transposed) weights [N][K].
// lane<16: N=lane, K={0..7}u{16..23}; lane>=16: N=lane-16, K={8..15}u{24..31}.
__device__ __forceinline__ v16h load_b_tile(const _Float16* __restrict__ base,
                                            int stride, int lane) {
  const int c    = lane & 15;
  const int half = lane >> 4;
  const _Float16* p = base + c * stride + half * 8;
  H16 u;
  u.h[0] = *(const v8h*)(p);
  u.h[1] = *(const v8h*)(p + 16);
  return u.v;
}

// One MLP layer as a tile loop. in: LDS activations [kM][sIn] f16.
// wt: global f16 transposed weights [N][K]. out: LDS f16 or f32.
template <bool LEAKY, bool F16OUT>
__device__ __forceinline__ void gemm_layer(const _Float16* __restrict__ in, int sIn,
                                           const _Float16* __restrict__ wt, int K,
                                           const float* __restrict__ bias, int nBias,
                                           void* __restrict__ outp, int sOut,
                                           int mTiles, int nTiles, int wave, int lane) {
  const int numTiles = mTiles * nTiles;
  for (int tile = wave; tile < numTiles; tile += kWaves) {   // wave-uniform
    const int m0 = (tile / nTiles) * 16;
    const int n0 = (tile % nTiles) * 16;
    v8f acc = {};
    for (int k0 = 0; k0 < K; k0 += 32) {
      v16h a = load_a_tile(in + m0 * sIn + k0, sIn, lane);
      v16h b = load_b_tile(wt + n0 * K + k0, K, lane);
      acc = __builtin_amdgcn_wmma_f32_16x16x32_f16(
          /*neg_a=*/false, a, /*neg_b=*/false, b,
          /*c_mod=*/(short)0, acc, /*reuse_a=*/false, /*reuse_b=*/false);
    }
    const int n  = n0 + (lane & 15);
    const int mb = m0 + ((lane >> 4) << 3);
    const float bv = (n < nBias) ? bias[n] : 0.0f;
#pragma unroll
    for (int g = 0; g < 8; ++g) {
      float v = acc[g] + bv;
      if (LEAKY) v = (v >= 0.0f) ? v : 0.01f * v;
      if (F16OUT) ((_Float16*)outp)[(mb + g) * sOut + n] = (_Float16)v;
      else        ((float*)outp)[(mb + g) * sOut + n] = v;
    }
  }
}

// Rigid-body derivatives for one row (M moments are zero in this model).
__device__ __forceinline__ void derivs(const float* __restrict__ x,
                                       const float* __restrict__ F,
                                       const float* __restrict__ I,
                                       const float* __restrict__ Ii,
                                       float* __restrict__ dx) {
  const float V0 = x[3], V1 = x[4], V2 = x[5];
  const float phi = x[6], th = x[7], psi = x[8];
  const float w0 = x[9], w1 = x[10], w2 = x[11];
  float sph, cph, sth, cth, sps, cps;
  sincosf(phi, &sph, &cph);
  sincosf(th,  &sth, &cth);
  sincosf(psi, &sps, &cps);
  // dpos = L_EB(phi,theta,psi) @ V
  dx[0] = (cth * cps) * V0 + (sph * sth * cps - cph * sps) * V1 + (cph * sth * cps + sph * sps) * V2;
  dx[1] = (cth * sps) * V0 + (sph * sth * sps + cph * cps) * V1 + (cph * sth * sps - sph * cps) * V2;
  dx[2] = -sth * V0 + sph * cth * V1 + cph * cth * V2;
  // dvel = F/m - omega x V
  dx[3] = F[0] * kInvMass - (w1 * V2 - w2 * V1);
  dx[4] = F[1] * kInvMass - (w2 * V0 - w0 * V2);
  dx[5] = F[2] * kInvMass - (w0 * V1 - w1 * V0);
  // datt = J(phi,theta) @ omega
  const float cthc = fmaxf(cth, kEps);
  const float tth  = tanf(th);
  dx[6] = w0 + (sph * w1 + cph * w2) * tth;
  dx[7] = cph * w1 - sph * w2;
  dx[8] = (sph * w1 + cph * w2) / cthc;
  // domega = I_inv @ (-(omega x (I @ omega)))
  const float Iw0 = I[0] * w0 + I[1] * w1 + I[2] * w2;
  const float Iw1 = I[3] * w0 + I[4] * w1 + I[5] * w2;
  const float Iw2 = I[6] * w0 + I[7] * w1 + I[8] * w2;
  const float c0 = w1 * Iw2 - w2 * Iw1;
  const float c1 = w2 * Iw0 - w0 * Iw2;
  const float c2 = w0 * Iw1 - w1 * Iw0;
  dx[9]  = -(Ii[0] * c0 + Ii[1] * c1 + Ii[2] * c2);
  dx[10] = -(Ii[3] * c0 + Ii[4] * c1 + Ii[5] * c2);
  dx[11] = -(Ii[6] * c0 + Ii[7] * c1 + Ii[8] * c2);
}

// Convert weights to f16, transposed to [N][K] (zero-padded).
__global__ void wm_prep(const float* __restrict__ W0, const float* __restrict__ W1,
                        const float* __restrict__ W2, const float* __restrict__ W3,
                        _Float16* __restrict__ w0t, _Float16* __restrict__ w1t,
                        _Float16* __restrict__ w2t, _Float16* __restrict__ w3t) {
  const int i = blockIdx.x * blockDim.x + threadIdx.x;
  if (i < 512 * 32) {               // W0: (16,512) -> [512][32], K pad
    const int n = i >> 5, k = i & 31;
    w0t[i] = (k < 16) ? (_Float16)W0[k * 512 + n] : (_Float16)0.0f;
  }
  if (i < 512 * 512) {              // W1/W2: (512,512) -> [512][512]
    const int n = i >> 9, k = i & 511;
    w1t[i] = (_Float16)W1[k * 512 + n];
    w2t[i] = (_Float16)W2[k * 512 + n];
  }
  if (i < 16 * 512) {               // W3: (512,6) -> [16][512], N pad
    const int n = i >> 9, k = i & 511;
    w3t[i] = (n < 6) ? (_Float16)W3[k * 6 + n] : (_Float16)0.0f;
  }
}

__global__ void __launch_bounds__(256)
wm_main(const float* __restrict__ x_t, const float* __restrict__ act,
        const float* __restrict__ Iptr,
        const float* __restrict__ b0, const float* __restrict__ b1,
        const float* __restrict__ b2, const float* __restrict__ b3,
        const _Float16* __restrict__ w0t, const _Float16* __restrict__ w1t,
        const _Float16* __restrict__ w2t, const _Float16* __restrict__ w3t,
        float* __restrict__ outF, float* __restrict__ outS) {
  extern __shared__ char smem_raw[];
  _Float16* inBuf = (_Float16*)smem_raw;            // [kM][32]
  _Float16* bufA  = inBuf + kInBufH;                // [kM][512]
  _Float16* bufB  = bufA + kBufH;                   // [kM][512]
  float*    fnBuf = (float*)(bufB + kBufH);         // [kM][16]
  float*    xS    = fnBuf + kFnF;                   // [kM][12]

  const int tid = threadIdx.x;
  const int wave = tid >> 5;
  const int lane = tid & 31;
  const int rowBase = blockIdx.x * kM;

  float I[9], Ii[9];
  if (tid < kM) {
    const int r = tid;
    const int gb = rowBase + r;
    // persistent state + stacked[:, :, 0]
    for (int c = 0; c < kState; ++c) {
      const float v = x_t[gb * kState + c];
      xS[r * kState + c] = v;
      outS[(size_t)gb * kState * kT + (size_t)c * kT + 0] = v;
    }
    for (int c = 16; c < 32; ++c) inBuf[r * 32 + c] = (_Float16)0.0f;
    // inertia + inverse (3x3 adjugate)
    for (int c = 0; c < 9; ++c) I[c] = Iptr[c];
    const float a = I[0], b = I[1], cc = I[2], d = I[3], e = I[4], f = I[5],
                g = I[6], h = I[7], ii = I[8];
    const float A = e * ii - f * h, Bc = -(d * ii - f * g), C = d * h - e * g;
    const float id = 1.0f / (a * A + b * Bc + cc * C);
    Ii[0] = A * id;  Ii[1] = (cc * h - b * ii) * id; Ii[2] = (b * f - cc * e) * id;
    Ii[3] = Bc * id; Ii[4] = (a * ii - cc * g) * id; Ii[5] = (cc * d - a * f) * id;
    Ii[6] = C * id;  Ii[7] = (b * g - a * h) * id;   Ii[8] = (a * e - b * d) * id;
  }
  __syncthreads();

  for (int t = 0; t < kTF; ++t) {
    // ---- build normalized MLP input (cols 0..11 state, 12..15 act, 16..31 0)
    if (tid < kM) {
      const int r = tid;
      const int gb = rowBase + r;
      const float NORMC[6] = {1.27323954473516f, 0.5f, 1.27323954473516f,
                              0.1f, 0.0166666666667f, 0.0166666666667f};
      for (int c = 0; c < 6; ++c)
        inBuf[r * 32 + c] = (_Float16)(xS[r * kState + c] * NORMC[c]);
      for (int c = 6; c < 12; ++c) inBuf[r * 32 + c] = (_Float16)0.0f;
      for (int a2 = 0; a2 < kActD; ++a2) {
        const float av = act[(size_t)gb * kActD * kT + (size_t)a2 * kT + (t + 1)];
        inBuf[r * 32 + 12 + a2] = (_Float16)((av - 1500.0f) * 0.002f);
      }
    }
    __syncthreads();

    // ---- MLP: four WMMA GEMM layers (ping-pong activations in LDS)
    gemm_layer<true,  true >(inBuf, 32, w0t, 32, b0, kH, bufB, kH, kM / 16, kH / 16, wave, lane);
    __syncthreads();
    gemm_layer<true,  true >(bufB, kH, w1t, kH, b1, kH, bufA, kH, kM / 16, kH / 16, wave, lane);
    __syncthreads();
    gemm_layer<true,  true >(bufA, kH, w2t, kH, b2, kH, bufB, kH, kM / 16, kH / 16, wave, lane);
    __syncthreads();
    gemm_layer<false, false>(bufB, kH, w3t, kH, b3, 6, fnBuf, 16, kM / 16, 1, wave, lane);
    __syncthreads();

    // ---- forces + RK4 integration (one wave, one thread per row)
    if (tid < kM) {
      const int r = tid;
      const int gb = rowBase + r;
      float F[3];
      F[0] = fnBuf[r * 16 + 0] - 0.5f;
      F[1] = fnBuf[r * 16 + 1] * -0.4f - 10.0f;
      F[2] = fnBuf[r * 16 + 2] * 0.2f - 0.1f;
      const size_t fb = (size_t)gb * 6 * kTF;
      outF[fb + 0 * kTF + t] = F[0];
      outF[fb + 1 * kTF + t] = F[1];
      outF[fb + 2 * kTF + t] = F[2];
      outF[fb + 3 * kTF + t] = 0.0f;
      outF[fb + 4 * kTF + t] = 0.0f;
      outF[fb + 5 * kTF + t] = 0.0f;

      float x[12], k1[12], k2[12], k3[12], k4[12], xt2[12];
      for (int c = 0; c < kState; ++c) x[c] = xS[r * kState + c];
      derivs(x, F, I, Ii, k1);
      for (int c = 0; c < kState; ++c) xt2[c] = x[c] + 0.5f * kDT * k1[c];
      derivs(xt2, F, I, Ii, k2);
      for (int c = 0; c < kState; ++c) xt2[c] = x[c] + 0.5f * kDT * k2[c];
      derivs(xt2, F, I, Ii, k3);
      for (int c = 0; c < kState; ++c) xt2[c] = x[c] + kDT * k3[c];
      derivs(xt2, F, I, Ii, k4);
      for (int c = 0; c < kState; ++c) {
        x[c] += (kDT / 6.0f) * (k1[c] + 2.0f * k2[c] + 2.0f * k3[c] + k4[c]);
        xS[r * kState + c] = x[c];
        outS[(size_t)gb * kState * kT + (size_t)c * kT + (t + 1)] = x[c];
      }
    }
    __syncthreads();
  }
}

extern "C" void kernel_launch(void* const* d_in, const int* in_sizes, int n_in,
                              void* d_out, int out_size, void* d_ws, size_t ws_size,
                              hipStream_t stream) {
  const float* x_t = (const float*)d_in[0];
  const float* act = (const float*)d_in[1];
  const float* Im  = (const float*)d_in[2];
  // d_in[3] = seq_len (compile-time constant kT here)
  const float* W0 = (const float*)d_in[4];
  const float* b0 = (const float*)d_in[5];
  const float* W1 = (const float*)d_in[6];
  const float* b1 = (const float*)d_in[7];
  const float* W2 = (const float*)d_in[8];
  const float* b2 = (const float*)d_in[9];
  const float* W3 = (const float*)d_in[10];
  const float* b3 = (const float*)d_in[11];

  _Float16* ws  = (_Float16*)d_ws;
  _Float16* w0t = ws + kW0T;
  _Float16* w1t = ws + kW1T;
  _Float16* w2t = ws + kW2T;
  _Float16* w3t = ws + kW3T;

  wm_prep<<<dim3((512 * 512 + 255) / 256), dim3(256), 0, stream>>>(
      W0, W1, W2, W3, w0t, w1t, w2t, w3t);

  float* outF = (float*)d_out;
  float* outS = outF + (size_t)kB * 6 * kTF;

  (void)hipFuncSetAttribute((const void*)wm_main,
                            hipFuncAttributeMaxDynamicSharedMemorySize,
                            kSmemBytes);
  wm_main<<<dim3(kB / kM), dim3(256), kSmemBytes, stream>>>(
      x_t, act, Im, b0, b1, b2, b3, w0t, w1t, w2t, w3t, outF, outS);
}
